// DTNBlock_23295902613828
// MI455X (gfx1250) — compile-verified, matrix operand order checked by reading, not run on hardware
//
#include <hip/hip_runtime.h>
#include <math.h>

// DTN block for MI455X (gfx1250, wave32). Dense matmuls -> v_wmma_f32_16x16x32_f16
// (f16 staging in LDS, f32 accum). Tropical (max-plus) matmuls -> VALU with LDS
// tiling, double-buffered via GLOBAL_LOAD_ASYNC_TO_LDS when available.
#define BB 2
#define LL 1024
#define DDIM 512
#define HH 8
#define FF 2048
#define PP 8
#define EPSLN 1e-5f
#define TPAD 68   // f32 tile pad: float4-aligned (async b128) + bank stride 4

typedef __attribute__((ext_vector_type(16))) _Float16 v16h;
typedef __attribute__((ext_vector_type(8)))  _Float16 v8h;
typedef __attribute__((ext_vector_type(8)))  float    v8f;

#if __has_builtin(__builtin_amdgcn_global_load_async_to_lds_b128)
#define DTN_ASYNC 1
typedef int v4i_ __attribute__((vector_size(16)));
typedef __attribute__((address_space(1))) v4i_ g_v4i;
typedef __attribute__((address_space(3))) v4i_ l_v4i;
__device__ __forceinline__ void async_cp16(void* lds, const void* g) {
  // GLOBAL_LOAD_ASYNC_TO_LDS_B128: per-lane global addr -> per-lane LDS addr,
  // no VGPR data round-trip, tracked with ASYNCcnt.
  __builtin_amdgcn_global_load_async_to_lds_b128(
      (g_v4i*)(uintptr_t)g, (l_v4i*)(uintptr_t)lds, 0, 0);
}
__device__ __forceinline__ void dtn_wait_async() {
#if __has_builtin(__builtin_amdgcn_s_wait_asynccnt)
  __builtin_amdgcn_s_wait_asynccnt(0);
#else
  asm volatile("s_wait_asynccnt 0x0" ::: "memory");
#endif
}
#else
#define DTN_ASYNC 0
#endif

// ---- WMMA helpers -----------------------------------------------------------
// A-frag (16x32 f16, lane holds row M=lane&15): per ISA 7.12.2, lane's 16 halves
// are two contiguous 8-half chunks: K = [half*8, half*8+8) and K = [16+half*8, ...).
// Same index math serves B-frags when the tile is stored (N,K)-major.
__device__ __forceinline__ v16h frag_ld_f16(const _Float16* base, int ld) {
  int lane = threadIdx.x & 31;
  const _Float16* p = base + (lane & 15) * ld + ((lane >> 4) << 3);
  v8h lo = *(const v8h*)(p);
  v8h hi = *(const v8h*)(p + 16);
  return __builtin_shufflevector(lo, hi, 0,1,2,3,4,5,6,7,8,9,10,11,12,13,14,15);
}

__device__ __forceinline__ v8f wmma_f16(v16h a, v16h b, v8f c) {
  return __builtin_amdgcn_wmma_f32_16x16x32_f16(false, a, false, b, (short)0, c,
                                                false, false);
}

__device__ __forceinline__ float gelu_exact(float x) {
  return 0.5f * x * (1.0f + erff(x * 0.70710678118654752f));
}
__device__ __forceinline__ float sigmoidf_(float x) {
  return 1.0f / (1.0f + expf(-x));
}

// ---- LayerNorm: one 256-thread block per row (D=512) ------------------------
__global__ void __launch_bounds__(256)
dtn_ln_kernel(const float* __restrict__ x, const float* __restrict__ g,
              const float* __restrict__ b, float* __restrict__ out) {
  __shared__ float red[16];
  __shared__ float mb[2];
  int row = blockIdx.x;
  const float* xr = x + (size_t)row * DDIM;
  int t = threadIdx.x;
  float a0 = xr[t], a1 = xr[t + 256];
  float s = a0 + a1, sq = a0 * a0 + a1 * a1;
  for (int off = 16; off; off >>= 1) {
    s  += __shfl_down(s, off, 32);
    sq += __shfl_down(sq, off, 32);
  }
  if ((t & 31) == 0) { red[t >> 5] = s; red[(t >> 5) + 8] = sq; }
  __syncthreads();
  if (t == 0) {
    float S = 0.f, Q = 0.f;
    for (int i = 0; i < 8; ++i) { S += red[i]; Q += red[i + 8]; }
    float mu = S / DDIM;
    mb[0] = mu;
    mb[1] = rsqrtf(Q / DDIM - mu * mu + EPSLN);
  }
  __syncthreads();
  float mu = mb[0], rs = mb[1];
  float* orow = out + (size_t)row * DDIM;
  orow[t]       = (a0 - mu) * rs * g[t]       + b[t];
  orow[t + 256] = (a1 - mu) * rs * g[t + 256] + b[t + 256];
}

// ---- Tropical (max-plus) linear: out[n,o] = max_k(X[n,k]+W[o,k]) + bias[o] ---
// 32x64 output tile; thread t handles row n=t>>3, cols oc0=(t&7)+8j (stride-8 so
// concurrent lanes hit consecutive LDS rows: bank stride 4, conflict-free).
// Async path: double-buffered GLOBAL_LOAD_ASYNC_TO_LDS_B128 prefetch of the next
// K-tile overlapped with the 1024-op VALU max-plus loop on the current tile.
__global__ void __launch_bounds__(256)
dtn_trop_kernel(const float* __restrict__ X, const float* __restrict__ W,
                const float* __restrict__ bias, float* __restrict__ out,
                int Din, int Dout) {
  __shared__ __align__(16) float Hs[2][32 * TPAD];
  __shared__ __align__(16) float Ws[2][64 * TPAD];
  int n0  = blockIdx.x * 32;
  int o0g = blockIdx.y * 64;
  int t = threadIdx.x;
  int n = t >> 3;
  int oc = t & 7;
  float acc[8];
#pragma unroll
  for (int j = 0; j < 8; ++j) acc[j] = -INFINITY;

#if DTN_ASYNC
  const int nk = Din >> 6;
  int buf = 0;
  // issue tile kt into buffer bf (Hs: 512 b128 chunks, Ws: 1024)
#define DTN_TROP_ISSUE(bf, k0)                                                  \
  {                                                                             \
    _Pragma("unroll")                                                           \
    for (int q = 0; q < 2; ++q) {                                               \
      int i = t + q * 256; int r = i >> 4, c4 = (i & 15) << 2;                  \
      async_cp16(&Hs[bf][r * TPAD + c4], X + (size_t)(n0 + r) * Din + (k0) + c4);\
    }                                                                           \
    _Pragma("unroll")                                                           \
    for (int q = 0; q < 4; ++q) {                                               \
      int i = t + q * 256; int r = i >> 4, c4 = (i & 15) << 2;                  \
      async_cp16(&Ws[bf][r * TPAD + c4], W + (size_t)(o0g + r) * Din + (k0) + c4);\
    }                                                                           \
  }
  DTN_TROP_ISSUE(0, 0)
  for (int kt = 0; kt < nk; ++kt) {
    dtn_wait_async();
    __syncthreads();                       // tile kt resident; prev compute done
    if (kt + 1 < nk) DTN_TROP_ISSUE(buf ^ 1, (kt + 1) << 6)
    const float* hrow = &Hs[buf][n * TPAD];
    const float* wb   = &Ws[buf][0];
#pragma unroll 4
    for (int kk = 0; kk < 64; ++kk) {
      float hv = hrow[kk];
#pragma unroll
      for (int j = 0; j < 8; ++j)
        acc[j] = fmaxf(acc[j], hv + wb[(oc + 8 * j) * TPAD + kk]);
    }
    buf ^= 1;
  }
#undef DTN_TROP_ISSUE
#else
  for (int k0 = 0; k0 < Din; k0 += 64) {
#pragma unroll
    for (int q = 0; q < 2; ++q) {
      int i = t + q * 256; int r = i >> 4, c4 = (i & 15) << 2;
      *(float4*)&Hs[0][r * TPAD + c4] =
          *(const float4*)(X + (size_t)(n0 + r) * Din + k0 + c4);
    }
#pragma unroll
    for (int q = 0; q < 4; ++q) {
      int i = t + q * 256; int r = i >> 4, c4 = (i & 15) << 2;
      *(float4*)&Ws[0][r * TPAD + c4] =
          *(const float4*)(W + (size_t)(o0g + r) * Din + k0 + c4);
    }
    __syncthreads();
#pragma unroll 4
    for (int kk = 0; kk < 64; ++kk) {
      float hv = Hs[0][n * TPAD + kk];
#pragma unroll
      for (int j = 0; j < 8; ++j)
        acc[j] = fmaxf(acc[j], hv + Ws[0][(oc + 8 * j) * TPAD + kk]);
    }
    __syncthreads();
  }
#endif
#pragma unroll
  for (int j = 0; j < 8; ++j)
    out[(size_t)(n0 + n) * Dout + o0g + oc + 8 * j] = acc[j] + bias[o0g + oc + 8 * j];
}

// ---- Dense GEMM: out = A(N,K) * W(M,K)^T + bias [+ resid], WMMA f16 ---------
// 256 threads = 8 waves, 128x64 tile; wave w owns rows 16w..16w+15 x 64 cols.
template <int EPI>
__global__ void __launch_bounds__(256)
dtn_gemm_wmma(const float* __restrict__ A, const float* __restrict__ W,
              const float* __restrict__ bias, const float* __restrict__ resid,
              float* __restrict__ out, int K, int M) {
  __shared__ __align__(16) _Float16 As[128 * 40];
  __shared__ __align__(16) _Float16 Ws[64 * 40];
  int row0 = blockIdx.x * 128;
  int col0 = blockIdx.y * 64;
  int t = threadIdx.x;
  int wave = t >> 5, lane = t & 31;
  v8f acc[4] = {};
  for (int k0 = 0; k0 < K; k0 += 32) {
#pragma unroll
    for (int q = 0; q < 4; ++q) {                      // A: 128x32 as float4s
      int i = t + q * 256;
      int r = i >> 3, c4 = (i & 7) << 2;
      float4 v = *(const float4*)(A + (size_t)(row0 + r) * K + k0 + c4);
      _Float16* d = As + r * 40 + c4;
      d[0] = (_Float16)v.x; d[1] = (_Float16)v.y;
      d[2] = (_Float16)v.z; d[3] = (_Float16)v.w;
    }
#pragma unroll
    for (int q = 0; q < 2; ++q) {                      // W: 64x32 as float4s
      int i = t + q * 256;
      int r = i >> 3, c4 = (i & 7) << 2;
      float4 v = *(const float4*)(W + (size_t)(col0 + r) * K + k0 + c4);
      _Float16* d = Ws + r * 40 + c4;
      d[0] = (_Float16)v.x; d[1] = (_Float16)v.y;
      d[2] = (_Float16)v.z; d[3] = (_Float16)v.w;
    }
    __syncthreads();
    v16h a = frag_ld_f16(As + wave * 16 * 40, 40);
#pragma unroll
    for (int ct = 0; ct < 4; ++ct) {
      v16h bm = frag_ld_f16(Ws + ct * 16 * 40, 40);
      acc[ct] = wmma_f16(a, bm, acc[ct]);
    }
    __syncthreads();
  }
  int nn = lane & 15, hi = lane >> 4;
#pragma unroll
  for (int ct = 0; ct < 4; ++ct) {
    int col = col0 + ct * 16 + nn;
    float bc = bias[col];
#pragma unroll
    for (int r = 0; r < 8; ++r) {
      int row = row0 + wave * 16 + r + hi * 8;
      size_t idx = (size_t)row * M + col;
      float v = acc[ct][r] + bc;
      if (EPI == 1) v += resid[idx];
      out[idx] = v;
    }
  }
}

// ---- Attention gate: g[n,h] = sigmoid(h . sg_W[h] + sg_b[h]) ----------------
__global__ void __launch_bounds__(256)
dtn_sg_kernel(const float* __restrict__ Hd, const float* __restrict__ Wg,
              const float* __restrict__ bg, float* __restrict__ g) {
  int n = blockIdx.x;
  int t = threadIdx.x;
  int head = t >> 5, lane = t & 31;
  const float* hr = Hd + (size_t)n * DDIM;
  const float* wr = Wg + (size_t)head * DDIM;
  float s = 0.f;
#pragma unroll
  for (int q = 0; q < DDIM / 32; ++q) s += hr[lane + q * 32] * wr[lane + q * 32];
  for (int off = 16; off; off >>= 1) s += __shfl_down(s, off, 32);
  if (lane == 0) g[(size_t)n * HH + head] = sigmoidf_(s + bg[head]);
}

// ---- Scores: 64x64 tile per block (128 thr / 4 waves), per (b,h) ------------
// cls = Q K^T via WMMA from pre-converted f16 LDS tiles; trop = max-plus on VALU
// from the f32 tiles, in the WMMA C layout; blended with the gate and written
// (pre-softmax) straight into the attn slice of d_out.
__global__ void __launch_bounds__(128)
dtn_scores_kernel(const float* __restrict__ Q, const float* __restrict__ Kl,
                  const float* __restrict__ g, float* __restrict__ sout) {
  __shared__ float Qs[64 * TPAD];
  __shared__ float Ks[64 * TPAD];
  __shared__ __align__(16) _Float16 Qh[64 * 72];
  __shared__ __align__(16) _Float16 Kh[64 * 72];
  int bh = blockIdx.z;
  int b = bh >> 3, h = bh & 7;
  int i0 = blockIdx.y * 64, j0 = blockIdx.x * 64;
  int t = threadIdx.x;
  int wave = t >> 5, lane = t & 31;
#pragma unroll
  for (int q = 0; q < 8; ++q) {                        // two 64x64 tiles, float4
    int i = t + q * 128;
    int r = i >> 4, c = (i & 15) << 2;
    float4 q4 = *(const float4*)(Q  + (size_t)(b * LL + i0 + r) * DDIM + h * 64 + c);
    float4 k4 = *(const float4*)(Kl + (size_t)(b * LL + j0 + r) * DDIM + h * 64 + c);
    float* qd = Qs + r * TPAD + c;
    float* kd = Ks + r * TPAD + c;
    qd[0] = q4.x; qd[1] = q4.y; qd[2] = q4.z; qd[3] = q4.w;
    kd[0] = k4.x; kd[1] = k4.y; kd[2] = k4.z; kd[3] = k4.w;
    _Float16* qh = Qh + r * 72 + c;
    _Float16* kh = Kh + r * 72 + c;
    qh[0] = (_Float16)q4.x; qh[1] = (_Float16)q4.y;
    qh[2] = (_Float16)q4.z; qh[3] = (_Float16)q4.w;
    kh[0] = (_Float16)k4.x; kh[1] = (_Float16)k4.y;
    kh[2] = (_Float16)k4.z; kh[3] = (_Float16)k4.w;
  }
  __syncthreads();
  v8f acc[4] = {};
#pragma unroll
  for (int ks = 0; ks < 2; ++ks) {                     // DK=64 -> 2 k-steps
    v16h a = frag_ld_f16(Qh + wave * 16 * 72 + ks * 32, 72);
#pragma unroll
    for (int ct = 0; ct < 4; ++ct) {
      v16h bm = frag_ld_f16(Kh + ct * 16 * 72 + ks * 32, 72);
      acc[ct] = wmma_f16(a, bm, acc[ct]);
    }
  }
  int nn = lane & 15, hi = lane >> 4;
  float trop[8][4];
#pragma unroll
  for (int r = 0; r < 8; ++r)
#pragma unroll
    for (int ct = 0; ct < 4; ++ct) trop[r][ct] = -INFINITY;
  for (int r = 0; r < 8; ++r) {
    const float* qrow = Qs + (wave * 16 + hi * 8 + r) * TPAD;
#pragma unroll 4
    for (int k = 0; k < 64; ++k) {
      float qv = qrow[k];
#pragma unroll
      for (int ct = 0; ct < 4; ++ct)
        trop[r][ct] = fmaxf(trop[r][ct], qv + Ks[(ct * 16 + nn) * TPAD + k]);
    }
  }
  const float scale = 0.125f;                          // DK^-0.5
#pragma unroll
  for (int r = 0; r < 8; ++r) {
    int i = i0 + wave * 16 + hi * 8 + r;
    float gi = g[(size_t)(b * LL + i) * HH + h];
#pragma unroll
    for (int ct = 0; ct < 4; ++ct) {
      int j = j0 + ct * 16 + nn;
      float cls = acc[ct][r] * scale;
      float tr  = trop[r][ct] * scale;
      sout[((size_t)bh * LL + i) * LL + j] = gi * tr + (1.0f - gi) * cls;
    }
  }
}

// ---- Softmax (in-place on attn slice of d_out), per-head temperature --------
__global__ void __launch_bounds__(256)
dtn_softmax_kernel(float* __restrict__ s, const float* __restrict__ temp) {
  __shared__ float red[8];
  __shared__ float bcast;
  int row = blockIdx.x;                                 // (b*H + h)*L + i
  int h = (row >> 10) & 7;
  float* sr = s + (size_t)row * LL;
  int t = threadIdx.x;
  float v[4];
  float m = -INFINITY;
#pragma unroll
  for (int i = 0; i < 4; ++i) { v[i] = sr[t + i * 256]; m = fmaxf(m, v[i]); }
  for (int off = 16; off; off >>= 1) m = fmaxf(m, __shfl_down(m, off, 32));
  if ((t & 31) == 0) red[t >> 5] = m;
  __syncthreads();
  if (t == 0) {
    float M_ = red[0];
    for (int i = 1; i < 8; ++i) M_ = fmaxf(M_, red[i]);
    bcast = M_;
  }
  __syncthreads();
  m = bcast;
  float invT = 1.0f / temp[h];
  float sum = 0.f;
#pragma unroll
  for (int i = 0; i < 4; ++i) { v[i] = expf((v[i] - m) * invT); sum += v[i]; }
  for (int off = 16; off; off >>= 1) sum += __shfl_down(sum, off, 32);
  __syncthreads();
  if ((t & 31) == 0) red[t >> 5] = sum;
  __syncthreads();
  if (t == 0) {
    float S = 0.f;
    for (int i = 0; i < 8; ++i) S += red[i];
    bcast = 1.0f / S;
  }
  __syncthreads();
  float inv = bcast;
#pragma unroll
  for (int i = 0; i < 4; ++i) sr[t + i * 256] = v[i] * inv;
}

// ---- o = attn @ V, with split-heads transpose fused in (writes (B,L,D)) -----
__global__ void __launch_bounds__(256)
dtn_attn_v_kernel(const float* __restrict__ attn, const float* __restrict__ V,
                  float* __restrict__ operm) {
  __shared__ __align__(16) _Float16 As[128 * 40];
  __shared__ __align__(16) _Float16 Vt[64 * 40];
  int bh = blockIdx.y;
  int b = bh >> 3, h = bh & 7;
  int i0 = blockIdx.x * 128;
  const float* Ab = attn + (size_t)bh * LL * LL;
  int t = threadIdx.x;
  int wave = t >> 5, lane = t & 31;
  v8f acc[4] = {};
  for (int k0 = 0; k0 < LL; k0 += 32) {
#pragma unroll
    for (int q = 0; q < 4; ++q) {                      // attn 128x32 as float4
      int i = t + q * 256;
      int r = i >> 3, c4 = (i & 7) << 2;
      float4 v = *(const float4*)(Ab + (size_t)(i0 + r) * LL + k0 + c4);
      _Float16* d = As + r * 40 + c4;
      d[0] = (_Float16)v.x; d[1] = (_Float16)v.y;
      d[2] = (_Float16)v.z; d[3] = (_Float16)v.w;
    }
#pragma unroll
    for (int q = 0; q < 8; ++q) {                      // V 32x64 -> Vt[d][k]
      int i = t + q * 256;
      int k = i >> 6, dcol = i & 63;
      Vt[dcol * 40 + k] =
          (_Float16)V[(size_t)(b * LL + k0 + k) * DDIM + h * 64 + dcol];
    }
    __syncthreads();
    v16h a = frag_ld_f16(As + wave * 16 * 40, 40);
#pragma unroll
    for (int ct = 0; ct < 4; ++ct) {
      v16h bm = frag_ld_f16(Vt + ct * 16 * 40, 40);
      acc[ct] = wmma_f16(a, bm, acc[ct]);
    }
    __syncthreads();
  }
  int nn = lane & 15, hi = lane >> 4;
#pragma unroll
  for (int ct = 0; ct < 4; ++ct) {
    int dcol = ct * 16 + nn;
#pragma unroll
    for (int r = 0; r < 8; ++r) {
      int i = i0 + wave * 16 + r + hi * 8;
      operm[(size_t)(b * LL + i) * DDIM + h * 64 + dcol] = acc[ct][r];
    }
  }
}

// ---- FFN blend: maxout PWL + GELU blend, gated fuse (in-place on tu) --------
__global__ void __launch_bounds__(256)
dtn_ffn_blend_kernel(float* __restrict__ tu, const float* __restrict__ cu,
                     const float* __restrict__ fg,
                     const float* __restrict__ slopes,
                     const float* __restrict__ inter,
                     const float* __restrict__ blend) {
  size_t idx = (size_t)blockIdx.x * 256 + threadIdx.x;
  int c = (int)(idx & (FF - 1));
  float x = tu[idx];
  float mx = -INFINITY;
#pragma unroll
  for (int p = 0; p < PP; ++p)
    mx = fmaxf(mx, slopes[c * PP + p] * x + inter[c * PP + p]);
  float sb    = sigmoidf_(blend[c]);
  float tropf = sb * mx + (1.0f - sb) * gelu_exact(x);
  float clsf  = gelu_exact(cu[idx]);
  float gf    = sigmoidf_(fg[idx]);
  tu[idx] = gf * tropf + (1.0f - gf) * clsf;
}

// ---- Launch -----------------------------------------------------------------
extern "C" void kernel_launch(void* const* d_in, const int* in_sizes, int n_in,
                              void* d_out, int out_size, void* d_ws, size_t ws_size,
                              hipStream_t stream) {
  const float* x      = (const float*)d_in[0];
  const float* ln1_g  = (const float*)d_in[1];
  const float* ln1_b  = (const float*)d_in[2];
  const float* q_W    = (const float*)d_in[3];
  const float* q_b    = (const float*)d_in[4];
  const float* k_W    = (const float*)d_in[5];
  const float* k_b    = (const float*)d_in[6];
  const float* v_W    = (const float*)d_in[7];
  const float* v_b    = (const float*)d_in[8];
  const float* out_W  = (const float*)d_in[9];
  const float* out_b  = (const float*)d_in[10];
  const float* sg_W   = (const float*)d_in[11];
  const float* sg_b   = (const float*)d_in[12];
  const float* temp   = (const float*)d_in[13];
  const float* ln2_g  = (const float*)d_in[14];
  const float* ln2_b  = (const float*)d_in[15];
  const float* tu_W   = (const float*)d_in[16];
  const float* tu_b   = (const float*)d_in[17];
  const float* lf_sl  = (const float*)d_in[18];
  const float* lf_in  = (const float*)d_in[19];
  const float* lf_bl  = (const float*)d_in[20];
  const float* cu_W   = (const float*)d_in[21];
  const float* cu_b   = (const float*)d_in[22];
  const float* fg_W   = (const float*)d_in[23];
  const float* fg_b   = (const float*)d_in[24];
  const float* down_W = (const float*)d_in[25];
  const float* down_b = (const float*)d_in[26];

  float* yout = (float*)d_out;                          // (B,L,D)
  float* attn = yout + (size_t)BB * LL * DDIM;          // (B,H,L,L)

  const size_t NROW = (size_t)BB * LL;                  // 2048
  float* ws    = (float*)d_ws;
  float* hbuf  = ws;                                    // (N,D)  h1 then h2
  float* qbuf  = hbuf + NROW * DDIM;                    // (N,D)  -> reused as operm
  float* kbuf  = qbuf + NROW * DDIM;                    // (N,D)  -> reused as y
  float* vbuf  = kbuf + NROW * DDIM;                    // (N,D)
  float* gbuf  = vbuf + NROW * DDIM;                    // (N,H)
  float* tubuf = gbuf + NROW * HH;                      // (N,F)  -> fused in place
  float* cubuf = tubuf + NROW * FF;                     // (N,F)
  float* fgbuf = cubuf + NROW * FF;                     // (N,F)
  float* operm = qbuf;                                  // safe: Q dead after scores
  float* ybuf  = kbuf;                                  // safe: K dead after scores

  dim3 b256(256);
  // attention sub-layer
  dtn_ln_kernel<<<(unsigned)NROW, b256, 0, stream>>>(x, ln1_g, ln1_b, hbuf);
  dtn_trop_kernel<<<dim3(NROW / 32, DDIM / 64), b256, 0, stream>>>(hbuf, q_W, q_b, qbuf, DDIM, DDIM);
  dtn_trop_kernel<<<dim3(NROW / 32, DDIM / 64), b256, 0, stream>>>(hbuf, k_W, k_b, kbuf, DDIM, DDIM);
  dtn_gemm_wmma<0><<<dim3(NROW / 128, DDIM / 64), b256, 0, stream>>>(hbuf, v_W, v_b, nullptr, vbuf, DDIM, DDIM);
  dtn_sg_kernel<<<(unsigned)NROW, b256, 0, stream>>>(hbuf, sg_W, sg_b, gbuf);
  dtn_scores_kernel<<<dim3(LL / 64, LL / 64, BB * HH), dim3(128), 0, stream>>>(qbuf, kbuf, gbuf, attn);
  dtn_softmax_kernel<<<BB * HH * LL, b256, 0, stream>>>(attn, temp);
  dtn_attn_v_kernel<<<dim3(LL / 128, BB * HH), b256, 0, stream>>>(attn, vbuf, operm);
  dtn_gemm_wmma<1><<<dim3(NROW / 128, DDIM / 64), b256, 0, stream>>>(operm, out_W, out_b, x, ybuf, DDIM, DDIM);
  // FFN sub-layer
  dtn_ln_kernel<<<(unsigned)NROW, b256, 0, stream>>>(ybuf, ln2_g, ln2_b, hbuf);
  dtn_trop_kernel<<<dim3(NROW / 32, FF / 64), b256, 0, stream>>>(hbuf, tu_W, tu_b, tubuf, DDIM, FF);
  dtn_gemm_wmma<0><<<dim3(NROW / 128, FF / 64), b256, 0, stream>>>(hbuf, cu_W, cu_b, nullptr, cubuf, DDIM, FF);
  dtn_gemm_wmma<0><<<dim3(NROW / 128, FF / 64), b256, 0, stream>>>(hbuf, fg_W, fg_b, nullptr, fgbuf, DDIM, FF);
  dtn_ffn_blend_kernel<<<(unsigned)(NROW * FF / 256), b256, 0, stream>>>(tubuf, cubuf, fgbuf, lf_sl, lf_in, lf_bl);
  dtn_gemm_wmma<1><<<dim3(NROW / 128, DDIM / 64), b256, 0, stream>>>(tubuf, down_W, down_b, ybuf, yout, FF, DDIM);
}
